// GCN300_51488067944594
// MI455X (gfx1250) — matrix operand order repaired and checked by reading, not compiled, not run on hardware
//
#include <hip/hip_runtime.h>
#include <hip/hip_bf16.h>

typedef __bf16 bf16_t;
typedef __attribute__((ext_vector_type(16))) __bf16 v16bf;
typedef __attribute__((ext_vector_type(8)))  __bf16 v8bf;
typedef __attribute__((ext_vector_type(8)))  float  v8f;

#define BN_EPS 1e-5f

// ---------------------------------------------------------------------------
// Degree / normalization
// ---------------------------------------------------------------------------
__global__ void gcn_deg_accum(const int* __restrict__ dst, float* __restrict__ deg, int E) {
    int i = blockIdx.x * blockDim.x + threadIdx.x;
    if (i < E) atomicAdd(&deg[dst[i]], 1.0f);
}

__global__ void gcn_dinv(float* __restrict__ deg, int N) {
    int i = blockIdx.x * blockDim.x + threadIdx.x;
    if (i < N) deg[i] = rsqrtf(deg[i] + 1.0f);   // self-loop: deg + 1
}

// ---------------------------------------------------------------------------
// fp32 -> bf16 row-major convert with column zero-padding
// ---------------------------------------------------------------------------
__global__ void gcn_cvt_pad(bf16_t* __restrict__ dstp, const float* __restrict__ srcp,
                            int C, int ldd, long total) {
    long i = (long)blockIdx.x * blockDim.x + threadIdx.x;
    if (i >= total) return;
    int r = (int)(i / ldd), c = (int)(i % ldd);
    dstp[i] = (bf16_t)(c < C ? srcp[(size_t)r * C + c] : 0.0f);
}

// W [Kvalid, Ncols] fp32 row-major -> WT [Npad, Kpad] bf16 (B-fragment friendly),
// zero outside valid region.
__global__ void gcn_wtrans(bf16_t* __restrict__ dstp, const float* __restrict__ W,
                           int Kvalid, int Ncols, int Kpad, long total) {
    long i = (long)blockIdx.x * blockDim.x + threadIdx.x;
    if (i >= total) return;
    int nn = (int)(i / Kpad), k = (int)(i % Kpad);
    float v = (nn < Ncols && k < Kvalid) ? W[(size_t)k * Ncols + nn] : 0.0f;
    dstp[i] = (bf16_t)v;
}

// Fold linear bias into BN affine: y = alpha * (xW) + beta
__global__ void gcn_bnparam(float* __restrict__ alpha, float* __restrict__ beta,
                            const float* __restrict__ g, const float* __restrict__ be,
                            const float* __restrict__ m, const float* __restrict__ v,
                            const float* __restrict__ b, int C) {
    int i = blockIdx.x * blockDim.x + threadIdx.x;
    if (i >= C) return;
    float a = g[i] * rsqrtf(v[i] + BN_EPS);
    alpha[i] = a;
    beta[i]  = be[i] + (b[i] - m[i]) * a;
}

// ---------------------------------------------------------------------------
// Register-blocked WMMA bf16 GEMM: C[M, Npad] = A[M, K] * B[K, Npad]
//   A row-major bf16 (lda = K); B supplied transposed as BT[Npad, K] bf16.
//   Each wave computes an MB x NB grid of 16x16 tiles: A fragments are reused
//   NB times, B fragments MB times (raises FLOP/byte ~2.6x vs 1x1 tiles).
//   K consumed 32/iter via v_wmma_f32_16x16x32_bf16.
//   Epilogue: optional BN affine + ReLU, masks columns >= Nvalid, writes
//   fp32 or bf16. Fragment layouts per CDNA5 ISA 7.12.2.
// ---------------------------------------------------------------------------
template<int MB, int NB>
__global__ __launch_bounds__(256)
void gcn_wmma_gemm(const bf16_t* __restrict__ A, int lda,
                   const bf16_t* __restrict__ BT, int ldb,
                   int K, int Mtiles, int Nvalid,
                   const float* __restrict__ bnAl, const float* __restrict__ bnBe,
                   int relu,
                   float* __restrict__ outF, bf16_t* __restrict__ outH, int ldc)
{
    const int lane = threadIdx.x & 31;
    const int wave = threadIdx.x >> 5;
    const int tM0  = (blockIdx.y * 8 + wave) * MB;
    const int tN0  = blockIdx.x * NB;
    if (tM0 >= Mtiles) return;            // wave-uniform: EXEC stays all-ones

    const int half = lane >> 4;           // K-half select (ISA lane split)
    const int l15  = lane & 15;

    // A: lane half 0 -> K {0..7, 16..23}; half 1 -> K {8..15, 24..31}
    size_t arow[MB];
    #pragma unroll
    for (int mi = 0; mi < MB; ++mi)
        arow[mi] = (size_t)((tM0 + mi) * 16 + l15) * lda + (size_t)half * 8;
    // B: lane half 0 -> K {0..7, 16..23}; half 1 -> K {8..15, 24..31}; row = output column
    size_t brow[NB];
    #pragma unroll
    for (int ni = 0; ni < NB; ++ni)
        brow[ni] = (size_t)((tN0 + ni) * 16 + l15) * ldb + (size_t)half * 8;

    v8f c[MB][NB] = {};

    for (int k0 = 0; k0 < K; k0 += 32) {
        v16bf a[MB], b[NB];
        #pragma unroll
        for (int mi = 0; mi < MB; ++mi) {
            const v8bf* Ap = (const v8bf*)(A + arow[mi] + k0);
            v8bf lo = Ap[0];              // K = k0 + half*8 + [0..7]
            v8bf hi = Ap[2];              // K = k0 + 16 + half*8 + [0..7]
            #pragma unroll
            for (int i = 0; i < 8; ++i) { a[mi][i] = lo[i]; a[mi][i + 8] = hi[i]; }
            if (k0 + 32 < K)
                __builtin_prefetch(A + arow[mi] + k0 + 32, 0, 1);  // global_prefetch_b8
        }
        #pragma unroll
        for (int ni = 0; ni < NB; ++ni) {
            const v8bf* Bp = (const v8bf*)(BT + brow[ni] + k0);
            v8bf lo = Bp[0];
            v8bf hi = Bp[2];
            #pragma unroll
            for (int i = 0; i < 8; ++i) { b[ni][i] = lo[i]; b[ni][i + 8] = hi[i]; }
        }
        #pragma unroll
        for (int mi = 0; mi < MB; ++mi)
            #pragma unroll
            for (int ni = 0; ni < NB; ++ni)
                // (neg_a, A, neg_b, B, c_mod, C, reuse_a, reuse_b)
                c[mi][ni] = __builtin_amdgcn_wmma_f32_16x16x32_bf16(
                    false, a[mi], false, b[ni], (short)0, c[mi][ni], false, false);
    }

    // Epilogue. C layout: VGPR v, lane -> (m = v + 8*half, n = l15).
    #pragma unroll
    for (int ni = 0; ni < NB; ++ni) {
        const int n  = (tN0 + ni) * 16 + l15;
        const bool nv = n < Nvalid;
        float alpha = 1.0f, beta = 0.0f;
        if (bnAl != nullptr && nv) { alpha = bnAl[n]; beta = bnBe[n]; }
        #pragma unroll
        for (int mi = 0; mi < MB; ++mi) {
            #pragma unroll
            for (int v = 0; v < 8; ++v) {
                int m = (tM0 + mi) * 16 + v + half * 8;
                float val = nv ? (c[mi][ni][v] * alpha + beta) : 0.0f;
                if (relu) val = fmaxf(val, 0.0f);
                if (outF) outF[(size_t)m * ldc + n] = val;
                else      outH[(size_t)m * ldc + n] = (bf16_t)val;
            }
        }
    }
}

// ---------------------------------------------------------------------------
// Edge scatter: agg[dst, f..f+3] += tmp[src, f..f+3] * dinv[src] * dinv[dst]
// (4 features per thread, vectorized load; F is always a multiple of 4)
// ---------------------------------------------------------------------------
__global__ void gcn_scatter(const float* __restrict__ tmp, int ldt,
                            float* __restrict__ agg, int F,
                            const int* __restrict__ src, const int* __restrict__ dst,
                            const float* __restrict__ dinv, long total) {
    long i = (long)blockIdx.x * blockDim.x + threadIdx.x;
    if (i >= total) return;
    const int Fq = F >> 2;
    int e = (int)(i / Fq), f = ((int)(i % Fq)) << 2;
    int s = src[e], d = dst[e];
    float coef = dinv[s] * dinv[d];
    const float4 t = *(const float4*)(tmp + (size_t)s * ldt + f);
    float* ap = agg + (size_t)d * F + f;
    atomicAdd(ap + 0, t.x * coef);
    atomicAdd(ap + 1, t.y * coef);
    atomicAdd(ap + 2, t.z * coef);
    atomicAdd(ap + 3, t.w * coef);
}

// Self-loop + bias + ReLU, emit next-layer bf16 activations (zero K-pad) or fp32.
__global__ void gcn_combine(const float* __restrict__ tmp, int ldt,
                            const float* __restrict__ agg, int F,
                            const float* __restrict__ dinv,
                            const float* __restrict__ bias,
                            bf16_t* __restrict__ outH, float* __restrict__ outF,
                            int ldo, long total) {
    long i = (long)blockIdx.x * blockDim.x + threadIdx.x;
    if (i >= total) return;
    int n = (int)(i / ldo), f = (int)(i % ldo);
    float val = 0.0f;
    if (f < F) {
        float di = dinv[n];
        val = agg[(size_t)n * F + f] + tmp[(size_t)n * ldt + f] * di * di + bias[f];
        val = fmaxf(val, 0.0f);
    }
    if (outH) outH[(size_t)n * ldo + f] = (bf16_t)val;
    else      outF[(size_t)n * ldo + f] = val;
}

// ---------------------------------------------------------------------------
// fc: out[g, j] = sum_k h[g*960 + k] * fcw[k, j] + fcb[j]   (G x 4 outputs)
// ---------------------------------------------------------------------------
__global__ void gcn_fc(const float* __restrict__ h, const float* __restrict__ fcw,
                       const float* __restrict__ fcb, float* __restrict__ out, int G) {
    int i = blockIdx.x * blockDim.x + threadIdx.x;
    if (i >= G * 4) return;
    int g = i >> 2, j = i & 3;
    const float* hr = h + (size_t)g * 960;
    float s = fcb[j];
    for (int k = 0; k < 960; ++k) s = fmaf(hr[k], fcw[k * 4 + j], s);
    out[i] = s;
}

// ---------------------------------------------------------------------------
// GEMM launch dispatch: pick widest N register blocking that divides Ntiles.
// ---------------------------------------------------------------------------
static void launch_gemm(const bf16_t* A, int lda, const bf16_t* BT, int ldb,
                        int K, int Mtiles, int Ntiles, int Nvalid,
                        const float* bnAl, const float* bnBe, int relu,
                        float* outF, bf16_t* outH, int ldc, hipStream_t stream)
{
    const dim3 blk(256);
    if (Ntiles % 4 == 0) {
        dim3 grid(Ntiles / 4, Mtiles / 16);
        gcn_wmma_gemm<2, 4><<<grid, blk, 0, stream>>>(A, lda, BT, ldb, K, Mtiles,
                                                      Nvalid, bnAl, bnBe, relu,
                                                      outF, outH, ldc);
    } else if (Ntiles % 2 == 0) {
        dim3 grid(Ntiles / 2, Mtiles / 16);
        gcn_wmma_gemm<2, 2><<<grid, blk, 0, stream>>>(A, lda, BT, ldb, K, Mtiles,
                                                      Nvalid, bnAl, bnBe, relu,
                                                      outF, outH, ldc);
    } else {
        dim3 grid(Ntiles, Mtiles / 16);
        gcn_wmma_gemm<2, 1><<<grid, blk, 0, stream>>>(A, lda, BT, ldb, K, Mtiles,
                                                      Nvalid, bnAl, bnBe, relu,
                                                      outF, outH, ldc);
    }
}

// ---------------------------------------------------------------------------
extern "C" void kernel_launch(void* const* d_in, const int* in_sizes, int n_in,
                              void* d_out, int out_size, void* d_ws, size_t ws_size,
                              hipStream_t stream)
{
    const float* x   = (const float*)d_in[0];
    const int*   ei  = (const int*)  d_in[1];
    const float* w1  = (const float*)d_in[2];
    const float* b1  = (const float*)d_in[3];
    const float* g1  = (const float*)d_in[4];
    const float* be1 = (const float*)d_in[5];
    const float* m1  = (const float*)d_in[6];
    const float* v1  = (const float*)d_in[7];
    const float* w2  = (const float*)d_in[8];
    const float* b2  = (const float*)d_in[9];
    const float* g2  = (const float*)d_in[10];
    const float* be2 = (const float*)d_in[11];
    const float* m2  = (const float*)d_in[12];
    const float* v2  = (const float*)d_in[13];
    const float* cw[5] = {(const float*)d_in[14], (const float*)d_in[16],
                          (const float*)d_in[18], (const float*)d_in[20],
                          (const float*)d_in[22]};
    const float* cb[5] = {(const float*)d_in[15], (const float*)d_in[17],
                          (const float*)d_in[19], (const float*)d_in[21],
                          (const float*)d_in[23]};
    const float* fcw = (const float*)d_in[24];
    const float* fcb = (const float*)d_in[25];

    const int N = in_sizes[0] / 250;     // 245760 (multiple of 256)
    const int E = in_sizes[1] / 2;       // 3932160
    const int* src = ei;
    const int* dst = ei + E;
    const int Mtiles = N / 16;           // 15360 (multiple of 16)
    const dim3 blk(256);

    // --- workspace partition ---
    char* ws = (char*)d_ws;
    size_t off = 0;
    auto wsAlloc = [&](size_t bytes) -> char* {
        char* p = ws + off;
        off += (bytes + 255) & ~(size_t)255;
        return p;
    };
    float*  dinv  = (float*) wsAlloc((size_t)N * 4);
    bf16_t* bufA  = (bf16_t*)wsAlloc((size_t)N * 1024 * 2);   // H1 (bf16, ld 1024)
    bf16_t* bufB  = (bf16_t*)wsAlloc((size_t)N * 256 * 2);    // xb / H2 / acts
    float*  tmp   = (float*) wsAlloc((size_t)N * 128 * 4);    // GEMM out pre-agg
    float*  agg   = (float*) wsAlloc((size_t)N * 128 * 4);    // scatter target
    float*  hlast = (float*) wsAlloc((size_t)N * 8 * 4);      // final [N,8] fp32
    bf16_t* wt    = (bf16_t*)wsAlloc((size_t)1024 * 1024 * 2);// transposed weights
    float*  bnAl  = (float*) wsAlloc(1024 * 4);
    float*  bnBe  = (float*) wsAlloc(1024 * 4);

    // --- degree / dinv ---
    hipMemsetAsync(dinv, 0, (size_t)N * 4, stream);
    gcn_deg_accum<<<(E + 255) / 256, blk, 0, stream>>>(dst, dinv, E);
    gcn_dinv<<<(N + 255) / 256, blk, 0, stream>>>(dinv, N);

    // --- x -> bf16 [N,256] in bufB ---
    {
        long t = (long)N * 256;
        gcn_cvt_pad<<<(t + 255) / 256, blk, 0, stream>>>(bufB, x, 250, 256, t);
    }

    // --- ffn1 layer 1: [N,256]x[256,1024v1000], BN1+ReLU -> bf16 bufA ld1024 ---
    {
        long t = (long)1024 * 256;
        gcn_wtrans<<<(t + 255) / 256, blk, 0, stream>>>(wt, w1, 250, 1000, 256, t);
        gcn_bnparam<<<(1000 + 255) / 256, blk, 0, stream>>>(bnAl, bnBe, g1, be1, m1, v1, b1, 1000);
        launch_gemm(bufB, 256, wt, 256, 256, Mtiles, 1024 / 16, 1000,
                    bnAl, bnBe, 1, nullptr, bufA, 1024, stream);
    }

    // --- ffn1 layer 2: [N,1024]x[1024,256v250], BN2 -> bf16 bufB ld256 ---
    {
        long t = (long)256 * 1024;
        gcn_wtrans<<<(t + 255) / 256, blk, 0, stream>>>(wt, w2, 1000, 250, 1024, t);
        gcn_bnparam<<<(250 + 255) / 256, blk, 0, stream>>>(bnAl, bnBe, g2, be2, m2, v2, b2, 250);
        launch_gemm(bufA, 1024, wt, 1024, 1024, Mtiles, 256 / 16, 250,
                    bnAl, bnBe, 0, nullptr, bufB, 256, stream);
    }

    // --- 5 GCN layers: 250->128->64->32->16->8 ---
    const int Kv[5]   = {250, 128, 64, 32, 16};   // valid K
    const int Kp[5]   = {256, 128, 64, 32, 32};   // padded K (mult of 32)
    const int Fo[5]   = {128, 64, 32, 16, 8};     // valid out features
    const int Np[5]   = {128, 64, 32, 16, 16};    // padded out (mult of 16)
    const int ldO[5]  = {128, 64, 32, 32, 8};     // next-layer activation stride
    bf16_t* actIn[5]  = {bufB, bufA, bufB, bufA, bufB};
    bf16_t* actOut[5] = {bufA, bufB, bufA, bufB, nullptr};

    for (int L = 0; L < 5; ++L) {
        long tw = (long)Np[L] * Kp[L];
        gcn_wtrans<<<(tw + 255) / 256, blk, 0, stream>>>(wt, cw[L], Kv[L], Fo[L], Kp[L], tw);

        launch_gemm(actIn[L], Kp[L], wt, Kp[L], Kp[L], Mtiles, Np[L] / 16, Fo[L],
                    nullptr, nullptr, 0, tmp, nullptr, Np[L], stream);

        hipMemsetAsync(agg, 0, (size_t)N * Fo[L] * 4, stream);
        long ts = (long)E * (Fo[L] / 4);
        gcn_scatter<<<(ts + 255) / 256, blk, 0, stream>>>(tmp, Np[L], agg, Fo[L],
                                                          src, dst, dinv, ts);
        long tc = (long)N * ldO[L];
        gcn_combine<<<(tc + 255) / 256, blk, 0, stream>>>(tmp, Np[L], agg, Fo[L], dinv,
                                                          cb[L], actOut[L],
                                                          (L == 4) ? hlast : nullptr,
                                                          ldO[L], tc);
    }

    // --- fc: [2048, 960] x [960, 4] + bias ---
    const int G = N / 120;
    gcn_fc<<<(G * 4 + 255) / 256, blk, 0, stream>>>(hlast, fcw, fcb, (float*)d_out, G);
}
